// YOLOLayer_16544214024749
// MI455X (gfx1250) — matrix operand order repaired and checked by reading, not compiled, not run on hardware
//
#include <hip/hip_runtime.h>
#include <cstdint>
#include <cstddef>

// ---------------------------------------------------------------------------
// Problem constants (from the reference)
// ---------------------------------------------------------------------------
#define BATCH   16
#define CIN     256
#define FDIM    76
#define HW      (FDIM*FDIM)        // 5776
#define OC      255                // A * (5 + 80)
#define NCH     85
#define NA      3
#define NLAB    40
#define NCELLS  (BATCH*NA*HW)      // 277248
#define KTILES  (CIN/32)           // 8
#define MTILES  16                 // 256 padded output channels / 16

typedef __attribute__((ext_vector_type(16))) __bf16 v16bf;
typedef __attribute__((ext_vector_type(8)))  float  v8f;

__device__ __forceinline__ float sigmoidf_(float v) { return 1.0f / (1.0f + __expf(-v)); }
__device__ __forceinline__ float bce1(float p, float t) {
    return -(t * fmaxf(__logf(p), -100.0f) + (1.0f - t) * fmaxf(__logf(1.0f - p), -100.0f));
}

// masked anchors (ANCHORS[:3] / 8)
__device__ __constant__ float c_maw[NA] = {1.25f, 2.0f,  4.125f};
__device__ __constant__ float c_mah[NA] = {1.625f, 3.75f, 2.875f};
// all anchors / 8
__device__ __constant__ float c_aw[9] = {1.25f, 2.0f, 4.125f, 3.75f, 7.75f, 7.375f, 14.5f, 19.5f, 46.625f};
__device__ __constant__ float c_ah[9] = {1.625f, 3.75f, 2.875f, 7.625f, 5.625f, 14.875f, 11.25f, 24.75f, 40.75f};

// ---------------------------------------------------------------------------
// 0) init: zero 6 loss slots, assignment table = -1
// ---------------------------------------------------------------------------
__global__ void init_kernel(int* __restrict__ assign_idx, float* __restrict__ dout) {
    int g = blockIdx.x * blockDim.x + threadIdx.x;
    if (g < NCELLS) assign_idx[g] = -1;
    if (g < 6)      dout[g] = 0.0f;
}

// ---------------------------------------------------------------------------
// 0b) repack W (fp32 [255][256]) into padded bf16 WMMA A-fragments:
//     layout [mt:16][kt:8][lane:32][elem:16] so the GEMM reads each lane's
//     fragment with one aligned 32B load. Row 255 padded with zeros.
//     A 16x32 bf16 per-lane layout used here (contiguous k, same order as B):
//       lanes 0-15 : M=lane,   K = 0..15
//       lanes 16-31: M=lane-16,K = 16..31
// ---------------------------------------------------------------------------
__global__ void prep_w_kernel(const float* __restrict__ Wm, __bf16* __restrict__ Wbf) {
    const int g = blockIdx.x * blockDim.x + threadIdx.x;   // 0 .. 16*8*32-1
    if (g >= MTILES * KTILES * 32) return;
    const int lane = g & 31;
    const int kt   = (g >> 5) & 7;
    const int mt   = g >> 8;
    const int o    = mt * 16 + (lane & 15);
    const int sel  = lane >> 4;

    v16bf frag;
    if (o < OC) {
        const float* wr = Wm + (size_t)o * CIN + kt * 32 + sel * 16;
        float4 a0 = *(const float4*)(wr + 0);
        float4 a1 = *(const float4*)(wr + 4);
        float4 a2 = *(const float4*)(wr + 8);
        float4 a3 = *(const float4*)(wr + 12);
        frag[0]  = (__bf16)a0.x; frag[1]  = (__bf16)a0.y;
        frag[2]  = (__bf16)a0.z; frag[3]  = (__bf16)a0.w;
        frag[4]  = (__bf16)a1.x; frag[5]  = (__bf16)a1.y;
        frag[6]  = (__bf16)a1.z; frag[7]  = (__bf16)a1.w;
        frag[8]  = (__bf16)a2.x; frag[9]  = (__bf16)a2.y;
        frag[10] = (__bf16)a2.z; frag[11] = (__bf16)a2.w;
        frag[12] = (__bf16)a3.x; frag[13] = (__bf16)a3.y;
        frag[14] = (__bf16)a3.z; frag[15] = (__bf16)a3.w;
    } else {
        #pragma unroll
        for (int i = 0; i < 16; ++i) frag[i] = (__bf16)0.0f;
    }
    *(v16bf*)(Wbf + (size_t)g * 16) = frag;
}

// ---------------------------------------------------------------------------
// 1) GEMM (W @ xin) + bias + selective sigmoid, bf16 WMMA, f32 accumulate.
//    Block = 128 threads (4 waves). Block tile: all 256 M (o) x 16 N (hw).
//    Wave w owns o in [w*64, w*64+64): 4 accumulators of 16x16.
//    xin K-tile staged in LDS as bf16 (register double-buffered);
//    W fragments read pre-packed from workspace (L2 resident, 128 KB).
// ---------------------------------------------------------------------------
__global__ __launch_bounds__(128) void gemm_act_kernel(
    const float* __restrict__ xin,    // [B][CIN][HW]
    const __bf16* __restrict__ Wbf,   // packed fragments [16][8][32][16]
    const float* __restrict__ bias,   // [OC]
    float* __restrict__ out)          // [B][OC][HW]
{
    __shared__ __attribute__((aligned(64))) __bf16 ldsB[32 * 16]; // [n][k] transposed

    const int tid   = threadIdx.x;
    const int wave  = tid >> 5;
    const int lane  = tid & 31;
    const int n15   = lane & 15;
    const int sel   = lane >> 4;          // half-wave select
    const int bb    = blockIdx.y;
    const int hwB   = blockIdx.x * 16;
    const int oW    = wave * 64;

    v8f acc[4];
    #pragma unroll
    for (int t = 0; t < 4; ++t)
        #pragma unroll
        for (int r = 0; r < 8; ++r) acc[t][r] = 0.0f;

    // cooperative-staging addressing (each thread: 4 consecutive n, one k)
    const int kl = tid >> 2;               // 0..31
    const int n0 = (tid & 3) * 4;          // 0,4,8,12
    const float* srcBase = xin + ((size_t)bb * CIN + kl) * HW + hwB + n0;

    // preload K-tile 0 into registers
    float4 cur = *(const float4*)srcBase;

    for (int kt = 0; kt < KTILES; ++kt) {
        if (kt > 0) __syncthreads();       // previous tile fully consumed
        ldsB[(n0 + 0) * 32 + kl] = (__bf16)cur.x;
        ldsB[(n0 + 1) * 32 + kl] = (__bf16)cur.y;
        ldsB[(n0 + 2) * 32 + kl] = (__bf16)cur.z;
        ldsB[(n0 + 3) * 32 + kl] = (__bf16)cur.w;
        __syncthreads();                   // tile visible

        // software pipeline: fetch next tile while WMMAs run on this one
        if (kt + 1 < KTILES)
            cur = *(const float4*)(srcBase + (size_t)(kt + 1) * 32 * HW);
        if (kt + 2 < KTILES)
            __builtin_prefetch(srcBase + (size_t)(kt + 2) * 32 * HW, 0, 0);

        // B fragment (32x16): lane n = lane&15; lanes 0-15 K=0..15, 16-31 K=16..31
        const v16bf bfrag = *(const v16bf*)&ldsB[n15 * 32 + sel * 16];

        // 4 A fragments, unconditional pre-packed loads (o=255 padded to zero)
        const __bf16* wp = Wbf + ((((size_t)(wave * 4) * KTILES + kt) * 32 + lane) * 16);
        #pragma unroll
        for (int t = 0; t < 4; ++t) {
            const v16bf afrag = *(const v16bf*)(wp + (size_t)t * KTILES * 32 * 16);
            acc[t] = __builtin_amdgcn_wmma_f32_16x16x32_bf16(
                false, afrag, false, bfrag, (short)0, acc[t], false, false);
        }
    }

    // epilogue: bias + sigmoid on ch<2 || ch>=4, store activation tensor
    #pragma unroll
    for (int t = 0; t < 4; ++t) {
        #pragma unroll
        for (int r = 0; r < 8; ++r) {
            const int o = oW + 16 * t + r + 8 * sel;   // D: lanes16-31 hold M=r+8
            if (o < OC) {
                float v = acc[t][r] + bias[o];
                const int ch = o % NCH;
                if (ch < 2 || ch >= 4) v = sigmoidf_(v);
                out[((size_t)bb * OC + o) * HW + hwB + n15] = v;
            }
        }
    }
}

// ---------------------------------------------------------------------------
// 2) per-label anchor matching -> assignment table
// ---------------------------------------------------------------------------
__global__ void label_kernel(const float* __restrict__ labels, int* __restrict__ assign_idx) {
    int g = blockIdx.x * blockDim.x + threadIdx.x;
    if (g >= BATCH * NLAB) return;
    const int b = g / NLAB;
    const float* L = labels + (size_t)g * 5;
    const float c0 = L[0], x = L[1], y = L[2], w = L[3], h = L[4];
    const bool valid = (c0 + x + y + w + h) > 0.0f;
    const float tx = x * FDIM, ty = y * FDIM, tw = w * FDIM, th = h * FDIM;
    const int   ti = (int)tx,  tj = (int)ty;

    // a_iou exactly as reference: tl=max((tx,ty),(0,0)); br=min((tw,th),(aw,ah))
    const float area_a = (tw - tx) * (th - ty);
    float best = -1e30f; int bi = 0;
    #pragma unroll
    for (int a = 0; a < 9; ++a) {
        const float tlx = fmaxf(tx, 0.0f), tly = fmaxf(ty, 0.0f);
        const float brx = fminf(tw, c_aw[a]), bry = fminf(th, c_ah[a]);
        const float inter = ((tlx < brx) && (tly < bry)) ? (brx - tlx) * (bry - tly) : 0.0f;
        float iou = inter / (area_a + c_aw[a] * c_ah[a] - inter);
        if (!valid) iou = -1.0f;
        if (iou > best) { best = iou; bi = a; }
    }
    if (valid && bi < NA) {
        const int lin = ((b * NA + bi) * FDIM + tj) * FDIM + ti;
        assign_idx[lin] = g;   // duplicate lin: one writer wins (matches scatter-set)
    }
}

// ---------------------------------------------------------------------------
// 3) per-cell loss accumulation
// ---------------------------------------------------------------------------
__global__ __launch_bounds__(256) void loss_kernel(
    const float* __restrict__ out,      // [B][OC][HW] activations
    const float* __restrict__ labels,
    const int*   __restrict__ assign_idx,
    float*       __restrict__ dout)     // 6 slots
{
    const int cell = blockIdx.x * 256 + threadIdx.x;
    float lxy = 0.f, lwh = 0.f, lobj = 0.f, lcls = 0.f, ll2 = 0.f;

    if (cell < NCELLS) {
        const int x = cell % FDIM;
        const int y = (cell / FDIM) % FDIM;
        const int a = (cell / HW) % NA;
        const int b = cell / (NA * HW);
        const int hw = y * FDIM + x;
        const size_t base = ((size_t)b * OC + a * NCH) * HW + hw;

        const float p0   = out[base];                 // sigmoided
        const float p1   = out[base + (size_t)HW];    // sigmoided
        const float r2   = out[base + (size_t)2 * HW];// raw
        const float r3   = out[base + (size_t)3 * HW];// raw
        const float pobj = out[base + (size_t)4 * HW];// sigmoided

        // pred box (stop_gradient values)
        const float px = p0 + (float)x, py = p1 + (float)y;
        const float pw = __expf(r2) * c_maw[a], ph = __expf(r3) * c_mah[a];
        const float ptlx = px - pw * 0.5f, ptly = py - ph * 0.5f;
        const float pbrx = px + pw * 0.5f, pbry = py + ph * 0.5f;

        // ignore mask: max IoU vs all valid labels of this batch
        float maxiou = 0.0f;
        const float* Lb = labels + (size_t)b * NLAB * 5;
        for (int k = 0; k < NLAB; ++k) {
            const float* L = Lb + k * 5;
            const float s = L[0] + L[1] + L[2] + L[3] + L[4];
            const float tx = L[1] * FDIM, ty = L[2] * FDIM;
            const float tw = L[3] * FDIM, th = L[4] * FDIM;
            const float ttlx = tx - tw * 0.5f, ttly = ty - th * 0.5f;
            const float tbrx = tx + tw * 0.5f, tbry = ty + th * 0.5f;
            const float ix0 = fmaxf(ptlx, ttlx), iy0 = fmaxf(ptly, ttly);
            const float ix1 = fminf(pbrx, tbrx), iy1 = fminf(pbry, tbry);
            const float inter = ((ix0 < ix1) && (iy0 < iy1)) ? (ix1 - ix0) * (iy1 - iy0) : 0.0f;
            const float denom = pw * ph + tw * th - inter;
            const float iou = (s > 0.0f) ? inter / denom : 0.0f;
            maxiou = fmaxf(maxiou, iou);
        }
        float objm = (maxiou > 0.7f) ? 0.0f : 1.0f;

        const int gl = assign_idx[cell];
        const bool assigned = (gl >= 0);
        if (assigned) objm = 1.0f;

        // objectness (all cells) + its l2 term
        const float tObj = assigned ? 1.0f : 0.0f;
        const float oObj = pobj * objm;
        lobj = bce1(oObj, tObj);
        ll2  = (oObj - tObj) * (oObj - tObj);

        if (assigned) {
            const float* L = labels + (size_t)gl * 5;
            const int cl = (int)L[0];
            const float tx = L[1] * FDIM, ty = L[2] * FDIM;
            const float tw = L[3] * FDIM, th = L[4] * FDIM;
            const int ti = (int)tx, tj = (int)ty;
            const float dx = tx - (float)ti, dy = ty - (float)tj;
            const float twl = __logf(tw / c_maw[a] + 1e-16f);
            const float thl = __logf(th / c_mah[a] + 1e-16f);
            const float scale = sqrtf(2.0f - tw * th / (float)(FDIM * FDIM));
            const float w2 = scale * scale;

            lxy = w2 * (bce1(p0, dx) + bce1(p1, dy));
            const float dw = (r2 - twl) * scale, dh = (r3 - thl) * scale;
            lwh = 0.5f * (dw * dw + dh * dh);               // LAMBDA folded in
            ll2 += (p0 - dx) * (p0 - dx) + (p1 - dy) * (p1 - dy) + dw * dw + dh * dh;

            for (int c = 0; c < 80; ++c) {
                const float p = out[base + (size_t)(5 + c) * HW]; // sigmoided
                const float t = (c == cl) ? 1.0f : 0.0f;
                lcls += bce1(p, t);
                ll2  += (p - t) * (p - t);
            }
        }
    }

    // block reduction of the 5 partials, then one atomic per slot
    __shared__ float red[256];
    float vals[5] = {lxy, lwh, lobj, lcls, ll2};
    #pragma unroll
    for (int i = 0; i < 5; ++i) {
        red[threadIdx.x] = vals[i];
        __syncthreads();
        for (int s = 128; s > 0; s >>= 1) {
            if ((int)threadIdx.x < s) red[threadIdx.x] += red[threadIdx.x + s];
            __syncthreads();
        }
        if (threadIdx.x == 0) atomicAdd(&dout[1 + i], red[0]);
        __syncthreads();
    }
}

// ---------------------------------------------------------------------------
// 4) total loss
// ---------------------------------------------------------------------------
__global__ void final_kernel(float* __restrict__ dout) {
    if (threadIdx.x == 0 && blockIdx.x == 0)
        dout[0] = dout[1] + dout[2] + dout[3] + dout[4];
}

// ---------------------------------------------------------------------------
// launch
// ---------------------------------------------------------------------------
extern "C" void kernel_launch(void* const* d_in, const int* in_sizes, int n_in,
                              void* d_out, int out_size, void* d_ws, size_t ws_size,
                              hipStream_t stream) {
    const float* xin    = (const float*)d_in[0];  // [16][256][76][76]
    const float* labels = (const float*)d_in[1];  // [16][40][5]
    const float* Wm     = (const float*)d_in[2];  // [255][256]
    const float* bias   = (const float*)d_in[3];  // [255]
    float* dout = (float*)d_out;                  // 6 floats

    const size_t actBytes    = (size_t)BATCH * OC * HW * sizeof(float); // ~94.3 MB
    const size_t assignBytes = (size_t)NCELLS * sizeof(int);            // ~1.1 MB
    float*  act    = (float*)d_ws;
    int*    assign = (int*)((char*)d_ws + actBytes);
    __bf16* Wbf    = (__bf16*)((char*)d_ws + actBytes + assignBytes);   // 128 KB

    init_kernel<<<(NCELLS + 255) / 256, 256, 0, stream>>>(assign, dout);

    prep_w_kernel<<<(MTILES * KTILES * 32 + 127) / 128, 128, 0, stream>>>(Wm, Wbf);

    dim3 gGrid(HW / 16, BATCH);   // (361, 16)
    gemm_act_kernel<<<gGrid, 128, 0, stream>>>(xin, Wbf, bias, act);

    label_kernel<<<(BATCH * NLAB + 63) / 64, 64, 0, stream>>>(labels, assign);

    loss_kernel<<<(NCELLS + 255) / 256, 256, 0, stream>>>(act, labels, assign, dout);

    final_kernel<<<1, 32, 0, stream>>>(dout);
}
